// EnsembleTransitionMLP_58385785422506
// MI455X (gfx1250) — matrix operand order repaired and checked by reading, compiled, not run on hardware
//
#include <hip/hip_runtime.h>

// ---------------------------------------------------------------------------
// EnsembleTransitionMLP for MI455X (gfx1250): fused 4-layer ensemble MLP.
// f16 WMMA (v_wmma_f32_16x16x32_f16) with f32 accumulation; activations fused
// through LDS; weights pre-converted/transposed to f16 K-major in d_ws; weight
// panels pulled into LDS by the Tensor Data Mover (tensor_load_to_lds) when
// the builtin exists, else cooperative b128 copies.
// ---------------------------------------------------------------------------

typedef __attribute__((ext_vector_type(16))) _Float16 v16h;
typedef __attribute__((ext_vector_type(8)))  _Float16 h8;
typedef __attribute__((ext_vector_type(8)))  float    v8f;
typedef __attribute__((ext_vector_type(4)))  unsigned int v4u;
typedef __attribute__((ext_vector_type(8)))  int          v8i32;
typedef __attribute__((ext_vector_type(4)))  int          v4i32;

#if defined(__AMDGCN__)
#if __has_builtin(__builtin_amdgcn_tensor_load_to_lds) && __has_builtin(__builtin_amdgcn_s_wait_tensorcnt)
#define USE_TDM 1
#endif
#endif
#ifndef USE_TDM
#define USE_TDM 0
#endif

// Problem sizes
constexpr int kB = 8192;   // batch
constexpr int kE = 50;     // ensemble
constexpr int kH = 256;    // hidden
// LDS layout: padded row stride 264 f16 (=528B, bank-rotating, 16B aligned)
#define HSTRIDE 264
#define WSTRIDE 264
static_assert(HSTRIDE % 8 == 0, "16B-aligned rows required");

// f16 workspace layout (element offsets), weights stored transposed K-major:
//  Wt[l][e][n][k]  (zero padded: layer1 K padded 40->64, layer4 N padded 33->64)
constexpr size_t OFF_W1 = 0;                                   // [E][256][64]
constexpr size_t OFF_W2 = OFF_W1 + (size_t)kE * 256 * 64;      // [E][256][256]
constexpr size_t OFF_W3 = OFF_W2 + (size_t)kE * 256 * 256;     // [E][256][256]
constexpr size_t OFF_W4 = OFF_W3 + (size_t)kE * 256 * 256;     // [E][64][256]
constexpr size_t WS_F16_ELEMS = OFF_W4 + (size_t)kE * 64 * 256; // ~16.4 MB

constexpr size_t REWARD_OFF = (size_t)kB * kE * 32;

constexpr unsigned WPANEL_F16_OFF  = 2u * 128u * HSTRIDE;           // after hA,hB
constexpr unsigned WPANEL_BYTE_OFF = WPANEL_F16_OFF * 2u;
constexpr unsigned SMEM_BYTES      = (2u * 128u * HSTRIDE + 64u * WSTRIDE) * 2u; // 168960 B

// ---------------------------------------------------------------------------
// transpose + f32->f16 convert:  out[e][n][k] = (k<K && n<N) ? in[e][k][n] : 0
// out is [E][NOUT][KOUT]; KOUT, NOUT multiples of 32.
// ---------------------------------------------------------------------------
__global__ void transcvt_kernel(const float* __restrict__ in, _Float16* __restrict__ out,
                                int K, int N, int KOUT, int NOUT) {
  __shared__ float tile[32][33];
  const int e  = blockIdx.z;
  const int k0 = blockIdx.x * 32;
  const int n0 = blockIdx.y * 32;
  const int tx = threadIdx.x, ty = threadIdx.y;   // 32 x 8
#pragma unroll
  for (int i = 0; i < 4; ++i) {
    const int k = k0 + ty + i * 8, n = n0 + tx;
    float v = 0.f;
    if (k < K && n < N) v = in[((size_t)e * K + k) * N + n];
    tile[ty + i * 8][tx] = v;
  }
  __syncthreads();
#pragma unroll
  for (int i = 0; i < 4; ++i) {
    const int n = n0 + ty + i * 8, k = k0 + tx;
    out[((size_t)e * NOUT + n) * KOUT + k] = (_Float16)tile[tx][ty + i * 8];
  }
}

// ---------------------------------------------------------------------------
// Load one 64-row x KDIM weight panel (contiguous f16 in global) into LDS with
// padded row stride WSTRIDE. TDM path: single descriptor, hardware inserts the
// 16B/row pad (pad_interval: 64q for K=256, 16q for K=64; pad_amount: 4 dw).
// ---------------------------------------------------------------------------
template <int KDIM>
__device__ __forceinline__ void loadPanel(_Float16* dstLds, const _Float16* src, int tid) {
#if USE_TDM
  (void)dstLds;
  if ((tid >> 5) == 0) {  // wave 0 only (wave-uniform branch, EXEC all ones)
    const unsigned long long ga = (unsigned long long)(uintptr_t)src;
    v4u g0;
    g0[0] = 1u;                                   // count=1, user descriptor
    g0[1] = WPANEL_BYTE_OFF;                      // lds_addr (bytes)
    g0[2] = (unsigned)(ga & 0xFFFFFFFFull);       // global_addr lo
    g0[3] = (unsigned)((ga >> 32) & 0x01FFFFFFull) | (2u << 30);  // ga hi | type=2
    constexpr unsigned padInterval = (KDIM == 256) ? 6u : 4u;     // 128/32 dwords
    v8i32 g1;
    g1[0] = (int)((1u << 16)            // data_size = 2 bytes
                  | (1u << 20)          // pad_enable
                  | (padInterval << 22) // pad_interval
                  | (3u << 25));        // pad_amount = 4 dwords (16B)
    g1[1] = (int)((unsigned)(KDIM & 0xFFFF) << 16);         // tensor_dim0[15:0]
    g1[2] = (int)(((unsigned)KDIM >> 16) | (64u << 16));    // tdim0 hi | tensor_dim1 lo
    g1[3] = (int)((unsigned)KDIM << 16);                    // tdim1 hi | tile_dim0
    g1[4] = 64;                                             // tile_dim1=64, tile_dim2=0
    g1[5] = KDIM;                                           // tensor_dim0_stride lo
    g1[6] = 0;
    g1[7] = 0;
    v4i32 z4 = {0, 0, 0, 0};
    v8i32 z8 = {0, 0, 0, 0, 0, 0, 0, 0};
    // 6-arg form (clang-23 / therock-10.0): (g0, g1, g2, g3, g4, cpol)
    __builtin_amdgcn_tensor_load_to_lds(g0, g1, z4, z4, z8, 0);
    __builtin_amdgcn_s_wait_tensorcnt(0);
  }
#else
  constexpr int CPR = KDIM / 8;  // 16B chunks per row
  for (int i = tid; i < 64 * CPR; i += 256) {
    const int row = i / CPR, ck = i - row * CPR;
    *(h8*)(dstLds + row * WSTRIDE + ck * 8) = *(const h8*)(src + (size_t)row * KDIM + ck * 8);
  }
#endif
}

// ---------------------------------------------------------------------------
// One dense layer: hOut[m][n] = act( sum_k hIn[m][k] * Wt[n][k] + bias[n] )
// M=128 rows per block; wave w owns rows [16w,16w+16). N streamed in panels
// of 64 (4 WMMA n-tiles, independent accumulator chains, k-outer/nt-inner).
// ---------------------------------------------------------------------------
template <int KDIM, int NTOT, bool RELU>
__device__ __forceinline__ void mlp_layer(const _Float16* __restrict__ wt,
                                          const float* __restrict__ bias,
                                          const _Float16* hIn, _Float16* hOut,
                                          _Float16* wP, int tid) {
  constexpr int KSTEPS = KDIM / 32;
  constexpr int NP     = NTOT / 64;
  const int wave = tid >> 5, lane = tid & 31;
  const int c = lane & 15, hs = lane >> 4;

  __syncthreads();  // hIn fully written (cross-wave for layer 1)

  // A fragments (16xKDIM, this wave's 16 rows), ISA 16-bit A layout:
  // lanes 0-15: K kb..kb+7 in v0-3, kb+16..23 in v4-7 (kb = hs*8)
  v16h A[KSTEPS];
#pragma unroll
  for (int k = 0; k < KSTEPS; ++k) {
    const _Float16* p = hIn + (wave * 16 + c) * HSTRIDE + k * 32 + hs * 8;
    ((h8*)&A[k])[0] = *(const h8*)p;
    ((h8*)&A[k])[1] = *(const h8*)(p + 16);
  }

  for (int np = 0; np < NP; ++np) {
    __syncthreads();                                    // panel consumed by all
    loadPanel<KDIM>(wP, wt + (size_t)np * 64 * KDIM, tid);
    __syncthreads();                                    // panel visible

    v8f acc[4] = {};
#pragma unroll
    for (int k = 0; k < KSTEPS; ++k) {
#pragma unroll
      for (int nt = 0; nt < 4; ++nt) {
        // B fragment (32x16): lane holds col c, K kb..kb+7 and kb+16..kb+23 (kb = hs*8)
        v16h Bf;
        const _Float16* p = wP + (nt * 16 + c) * WSTRIDE + k * 32 + hs * 8;
        ((h8*)&Bf)[0] = *(const h8*)p;
        ((h8*)&Bf)[1] = *(const h8*)(p + 16);
        acc[nt] = __builtin_amdgcn_wmma_f32_16x16x32_f16(
            false, A[k], false, Bf, (short)0, acc[nt], false, false);
      }
    }
    // epilogue: bias + relu + f16 store (C layout: lane->col c, v -> row hs*8+v)
#pragma unroll
    for (int nt = 0; nt < 4; ++nt) {
      const int nGlob = np * 64 + nt * 16 + c;
      const float bv = bias[nGlob];
#pragma unroll
      for (int v = 0; v < 8; ++v) {
        float x = acc[nt][v] + bv;
        if (RELU) x = fmaxf(x, 0.f);
        hOut[(wave * 16 + hs * 8 + v) * HSTRIDE + nGlob] = (_Float16)x;
      }
    }
  }
}

// ---------------------------------------------------------------------------
// Fused main kernel: block = (e, 128-row tile). 256 threads = 8 waves.
// ---------------------------------------------------------------------------
__global__ void __launch_bounds__(256)
ensemble_mlp_kernel(const float* __restrict__ state, const float* __restrict__ action,
                    const _Float16* __restrict__ ws,
                    const float* __restrict__ b1, const float* __restrict__ b2,
                    const float* __restrict__ b3, const float* __restrict__ b4,
                    float* __restrict__ out) {
  extern __shared__ _Float16 smem[];
  _Float16* hA = smem;                       // [128][HSTRIDE]
  _Float16* hB = smem + 128 * HSTRIDE;       // [128][HSTRIDE]
  _Float16* wP = smem + WPANEL_F16_OFF;      // [64][WSTRIDE]

  const int tid  = threadIdx.x;
  const int e    = blockIdx.x >> 6;          // 64 row-blocks per member
  const int rb   = blockIdx.x & 63;
  const int row0 = rb * 128;

  // Stage 0: sa = concat(state, action) -> hA[:, 0:64] f16, zero-padded 40..63
  for (int i = tid; i < 128 * 64; i += 256) {
    const int r = i >> 6, k = i & 63;
    float v = 0.f;
    if (k < 32)       v = state [(size_t)(row0 + r) * 32 + k];
    else if (k < 40)  v = action[(size_t)(row0 + r) * 8 + (k - 32)];
    hA[r * HSTRIDE + k] = (_Float16)v;
  }

  mlp_layer< 64, 256, true>(ws + OFF_W1 + (size_t)e * (256 * 64),  b1 + e * kH, hA, hB, wP, tid);
  mlp_layer<256, 256, true>(ws + OFF_W2 + (size_t)e * (256 * 256), b2 + e * kH, hB, hA, wP, tid);
  mlp_layer<256, 256, true>(ws + OFF_W3 + (size_t)e * (256 * 256), b3 + e * kH, hA, hB, wP, tid);

  // ---- layer 4: [128x256] x [256x33(->48)] -> global (next_state, reward) ----
  __syncthreads();
  loadPanel<256>(wP, ws + OFF_W4 + (size_t)e * (64 * 256), tid);
  __syncthreads();
  {
    const int wave = tid >> 5, lane = tid & 31;
    const int c = lane & 15, hs = lane >> 4;
    v16h A[8];
#pragma unroll
    for (int k = 0; k < 8; ++k) {
      const _Float16* p = hB + (wave * 16 + c) * HSTRIDE + k * 32 + hs * 8;
      ((h8*)&A[k])[0] = *(const h8*)p;
      ((h8*)&A[k])[1] = *(const h8*)(p + 16);
    }
    v8f acc[3] = {};
#pragma unroll
    for (int k = 0; k < 8; ++k) {
#pragma unroll
      for (int nt = 0; nt < 3; ++nt) {
        v16h Bf;
        const _Float16* p = wP + (nt * 16 + c) * WSTRIDE + k * 32 + hs * 8;
        ((h8*)&Bf)[0] = *(const h8*)p;
        ((h8*)&Bf)[1] = *(const h8*)(p + 16);
        acc[nt] = __builtin_amdgcn_wmma_f32_16x16x32_f16(
            false, A[k], false, Bf, (short)0, acc[nt], false, false);
      }
    }
#pragma unroll
    for (int nt = 0; nt < 3; ++nt) {
      const int col = nt * 16 + c;
      const float bv = (col < 33) ? b4[e * 33 + col] : 0.f;
#pragma unroll
      for (int v = 0; v < 8; ++v) {
        const float x = acc[nt][v] + bv;
        const int r = row0 + wave * 16 + hs * 8 + v;
        if (col < 32)        out[((size_t)r * kE + e) * 32 + col] = x;   // next_state
        else if (col == 32)  out[REWARD_OFF + (size_t)r * kE + e] = x;   // reward
      }
    }
  }
}

// ---------------------------------------------------------------------------
extern "C" void kernel_launch(void* const* d_in, const int* in_sizes, int n_in,
                              void* d_out, int out_size, void* d_ws, size_t ws_size,
                              hipStream_t stream) {
  (void)in_sizes; (void)n_in; (void)out_size; (void)ws_size;
  const float* state  = (const float*)d_in[0];
  const float* action = (const float*)d_in[1];
  const float* W1 = (const float*)d_in[2];
  const float* b1 = (const float*)d_in[3];
  const float* W2 = (const float*)d_in[4];
  const float* b2 = (const float*)d_in[5];
  const float* W3 = (const float*)d_in[6];
  const float* b3 = (const float*)d_in[7];
  const float* W4 = (const float*)d_in[8];
  const float* b4 = (const float*)d_in[9];
  _Float16* ws = (_Float16*)d_ws;            // needs >= WS_F16_ELEMS*2 bytes (~15.6 MB)
  float* out = (float*)d_out;

  const dim3 tb(32, 8, 1);
  // W1: [E][40][256] -> Wt1 [E][256][64]
  transcvt_kernel<<<dim3(64 / 32, 256 / 32, kE), tb, 0, stream>>>(W1, ws + OFF_W1, 40, 256, 64, 256);
  // W2/W3: [E][256][256] -> transposed [E][256][256]
  transcvt_kernel<<<dim3(256 / 32, 256 / 32, kE), tb, 0, stream>>>(W2, ws + OFF_W2, 256, 256, 256, 256);
  transcvt_kernel<<<dim3(256 / 32, 256 / 32, kE), tb, 0, stream>>>(W3, ws + OFF_W3, 256, 256, 256, 256);
  // W4: [E][256][33] -> Wt4 [E][64][256] (rows 33..63 zero)
  transcvt_kernel<<<dim3(256 / 32, 64 / 32, kE), tb, 0, stream>>>(W4, ws + OFF_W4, 256, 33, 256, 64);

  ensemble_mlp_kernel<<<dim3(kE * (kB / 128)), dim3(256), SMEM_BYTES, stream>>>(
      state, action, ws, b1, b2, b3, b4, out);
}
